// REIGNN_77326591197668
// MI455X (gfx1250) — matrix operand order repaired and checked by reading, not compiled, not run on hardware
//
#include <hip/hip_runtime.h>
#include <hip/hip_bf16.h>
#include <math.h>

// ---------------------------------------------------------------------------
// CDNA5 (gfx1250) implementation. wave32. Matrix ops via v_wmma_f32_16x16x32_f16.
// Weights pre-packed to f16 in WMMA fragment order (one 32B v16h per lane).
// ---------------------------------------------------------------------------

typedef __attribute__((ext_vector_type(16))) _Float16 v16h;
typedef __attribute__((ext_vector_type(8)))  float    v8f;

#define GF_RELU  1
#define GF_ACCUM 2
#define DH 128   // hidden dim

static __device__ __forceinline__ float sigmoidf_(float x) { return 1.0f / (1.0f + expf(-x)); }

// float atomic max via int punning (works for mixed signs)
static __device__ __forceinline__ void atomicMaxF(float* addr, float v) {
    if (v >= 0.0f) atomicMax((int*)addr, __float_as_int(v));
    else           atomicMin((unsigned int*)addr, __float_as_uint(v));
}

// ---------------------------------------------------------------------------
// Weight pack: f32 B -> f16 fragment-ordered buffer.
// Output layout: block (ks, t) of 32 lanes x 16 halfs, flat index
//   o = ((ks*(N/16) + t)*32 + lane)*16 + e
// holding B[k = ks*32 + (e<8 ? (lane>>4)*8 + e : 16 + (lane>>4)*8 + e-8)][n = t*16 + (lane&15)]  (per ISA
// 7.12.2 B-fragment layout).  transB: source is [N,Ksrc] used as B^T.
// k >= Ksrc zero-filled (K padding for the 147-wide pre-linear).
// ---------------------------------------------------------------------------
__global__ void pack_b_kernel(const float* __restrict__ B, _Float16* __restrict__ out,
                              int Kp, int Ksrc, int N, int transB) {
    size_t o = (size_t)blockIdx.x * blockDim.x + threadIdx.x;
    if (o >= (size_t)Kp * N) return;
    int e    = (int)(o & 15);
    int lane = (int)((o >> 4) & 31);
    int l = lane & 15, hi = lane >> 4;
    int nt = N >> 4;
    size_t blk = o >> 9;              // ks*nt + t
    int t  = (int)(blk % nt);
    int ks = (int)(blk / nt);
    int k = ks * 32 + (e < 8 ? hi * 8 + e : 16 + hi * 8 + (e - 8));
    int n = t * 16 + l;
    float val = 0.0f;
    if (k < Ksrc) val = transB ? B[(size_t)n * Ksrc + k] : B[(size_t)k * N + n];
    out[o] = (_Float16)val;
}

// ---------------------------------------------------------------------------
// WMMA GEMM: C[M,TN] = act( A[M,K] @ Bp (+ bias) (+ C) ), Bp packed f16.
// Requirements: K % 32 == 0 (M arbitrary; load rows clamped, stores guarded
// only in the tail tile). Block = 256 threads = 8 waves; wave w computes rows
// [blockIdx.x*128 + w*16, +16) x cols [blockIdx.y*64, +64) as 4 WMMA tiles:
// each A fragment feeds 4 v_wmma; each B fragment is one 32B contiguous load.
// ---------------------------------------------------------------------------
template <int TN>
__global__ __launch_bounds__(256)
void wmma_gemm_kernel(const float* __restrict__ A, const _Float16* __restrict__ Bp,
                      const float* __restrict__ bias, float* __restrict__ C,
                      int M, int K, int flags)
{
    const int wave = threadIdx.x >> 5;
    const int lane = threadIdx.x & 31;
    const int l    = lane & 15;
    const int hi   = lane >> 4;
    const int row0 = blockIdx.x * 128 + wave * 16;
    if (row0 >= M) return;                      // wave-uniform exit
    const int col0 = blockIdx.y * 64;
    const int t0   = col0 >> 4;                 // first B tile index

    const int arow = min(row0 + l, M - 1);      // clamp: loads always in-bounds
    const float* __restrict__ Arow = A + (size_t)arow * K;
    const v16h* __restrict__ Bv = (const v16h*)Bp;

    v8f acc[4] = {{}, {}, {}, {}};

#pragma unroll 2
    for (int k0 = 0; k0 < K; k0 += 32) {
        // ---- A fragment: lane (l,hi) holds row arow, K = k0+hi*8+{0..7} and +16
        const float4* pa0 = (const float4*)(Arow + k0 + hi * 8);
        const float4* pa1 = (const float4*)(Arow + k0 + 16 + hi * 8);
        float4 x0 = pa0[0], x1 = pa0[1], x2 = pa1[0], x3 = pa1[1];
        v16h a;
        a[0]  = (_Float16)x0.x; a[1]  = (_Float16)x0.y; a[2]  = (_Float16)x0.z; a[3]  = (_Float16)x0.w;
        a[4]  = (_Float16)x1.x; a[5]  = (_Float16)x1.y; a[6]  = (_Float16)x1.z; a[7]  = (_Float16)x1.w;
        a[8]  = (_Float16)x2.x; a[9]  = (_Float16)x2.y; a[10] = (_Float16)x2.z; a[11] = (_Float16)x2.w;
        a[12] = (_Float16)x3.x; a[13] = (_Float16)x3.y; a[14] = (_Float16)x3.z; a[15] = (_Float16)x3.w;

        const size_t kb = (size_t)(k0 >> 5) * (TN >> 4);
#pragma unroll
        for (int j = 0; j < 4; ++j) {
            v16h b = Bv[(kb + t0 + j) * 32 + lane];   // one 32B contiguous load
            acc[j] = __builtin_amdgcn_wmma_f32_16x16x32_f16(false, a, false, b,
                                                            (short)0, acc[j], false, false);
        }
    }

    // ---- epilogue: C 16x16 layout -> row r (lanes0-15) / r+8 (lanes16-31), col = lane%16
    if (row0 + 16 <= M) {           // fast path: full tile, no per-row guards
#pragma unroll
        for (int j = 0; j < 4; ++j) {
            const int ocol = col0 + j * 16 + l;
            const float bv = bias ? bias[ocol] : 0.0f;
#pragma unroll
            for (int r = 0; r < 8; ++r) {
                size_t idx = (size_t)(row0 + r + hi * 8) * TN + ocol;
                float v = acc[j][r];
                if (flags & GF_ACCUM) v += C[idx];
                v += bv;
                if (flags & GF_RELU)  v = fmaxf(v, 0.0f);
                C[idx] = v;
            }
        }
    } else {                        // tail tile: per-row guard
#pragma unroll
        for (int j = 0; j < 4; ++j) {
            const int ocol = col0 + j * 16 + l;
            const float bv = bias ? bias[ocol] : 0.0f;
#pragma unroll
            for (int r = 0; r < 8; ++r) {
                int orow = row0 + r + hi * 8;
                if (orow < M) {
                    size_t idx = (size_t)orow * TN + ocol;
                    float v = acc[j][r];
                    if (flags & GF_ACCUM) v += C[idx];
                    v += bv;
                    if (flags & GF_RELU)  v = fmaxf(v, 0.0f);
                    C[idx] = v;
                }
            }
        }
    }
}

// ---------------------------------------------------------------------------
// Elementwise / init kernels
// ---------------------------------------------------------------------------
__global__ void fill_kernel(float* __restrict__ p, float v, size_t n) {
    size_t i = (size_t)blockIdx.x * blockDim.x + threadIdx.x;
    if (i < n) p[i] = v;
}

__global__ void set_bias_rows_kernel(float* __restrict__ out, const float* __restrict__ bias,
                                     size_t rows, int d) {
    size_t i = (size_t)blockIdx.x * blockDim.x + threadIdx.x;
    if (i < rows * (size_t)d) out[i] = bias[i % d];
}

__global__ void relu_kernel(float* __restrict__ p, size_t n) {
    size_t i = (size_t)blockIdx.x * blockDim.x + threadIdx.x;
    if (i < n) { float v = p[i]; p[i] = v > 0.0f ? v : 0.0f; }
}

__global__ void add_kernel(const float* __restrict__ a, const float* __restrict__ b,
                           float* __restrict__ c, size_t n) {
    size_t i = (size_t)blockIdx.x * blockDim.x + threadIdx.x;
    if (i < n) c[i] = a[i] + b[i];
}

// write x_a|collab into zero-padded [Na, stride] activation buffer (first fa+d cols)
__global__ void concat_kernel(const float* __restrict__ xa, const float* __restrict__ collab,
                              float* __restrict__ out, int Na, int fa, int d, int stride) {
    size_t i = (size_t)blockIdx.x * blockDim.x + threadIdx.x;
    size_t tot = (size_t)Na * (fa + d);
    if (i >= tot) return;
    int row = (int)(i / (fa + d));
    int col = (int)(i % (fa + d));
    out[(size_t)row * stride + col] =
        (col < fa) ? xa[(size_t)row * fa + col] : collab[(size_t)row * d + (col - fa)];
}

// ---------------------------------------------------------------------------
// GATv2 edge kernels: one wave per edge, one float4 per lane (32*4 = 128 feats)
// ---------------------------------------------------------------------------
__global__ void edge_score_kernel(const float* __restrict__ xl, const float* __restrict__ xr,
                                  const float* __restrict__ att, const int* __restrict__ src,
                                  const int* __restrict__ dst, float* __restrict__ e, int E) {
    int gid = blockIdx.x * blockDim.x + threadIdx.x;
    int edge = gid >> 5, lane = gid & 31;
    if (edge >= E) return;
    float4 va = ((const float4*)(xl + (size_t)src[edge] * DH))[lane];
    float4 vb = ((const float4*)(xr + (size_t)dst[edge] * DH))[lane];
    float4 w  = ((const float4*)att)[lane];
    float s0 = va.x + vb.x, s1 = va.y + vb.y, s2 = va.z + vb.z, s3 = va.w + vb.w;
    s0 = s0 > 0.f ? s0 : 0.2f * s0;  s1 = s1 > 0.f ? s1 : 0.2f * s1;
    s2 = s2 > 0.f ? s2 : 0.2f * s2;  s3 = s3 > 0.f ? s3 : 0.2f * s3;
    float p = w.x * s0 + w.y * s1 + w.z * s2 + w.w * s3;
    for (int off = 16; off > 0; off >>= 1) p += __shfl_down(p, off, 32);
    if (lane == 0) e[edge] = p;
}

__global__ void edge_max_kernel(const float* __restrict__ e, const int* __restrict__ dst,
                                float* __restrict__ emax, int E) {
    int i = blockIdx.x * blockDim.x + threadIdx.x;
    if (i < E) atomicMaxF(&emax[dst[i]], e[i]);
}

__global__ void edge_exp_kernel(const float* __restrict__ e, const int* __restrict__ dst,
                                const float* __restrict__ emax, float* __restrict__ ee,
                                float* __restrict__ denom, int E) {
    int i = blockIdx.x * blockDim.x + threadIdx.x;
    if (i >= E) return;
    int d = dst[i];
    float x = expf(e[i] - emax[d]);
    ee[i] = x;
    atomicAdd(&denom[d], x);
}

__global__ void edge_scatter_kernel(const float* __restrict__ xl, const float* __restrict__ ee,
                                    const float* __restrict__ denom, const int* __restrict__ src,
                                    const int* __restrict__ dst, float* __restrict__ out, int E) {
    int gid = blockIdx.x * blockDim.x + threadIdx.x;
    int edge = gid >> 5, lane = gid & 31;
    if (edge >= E) return;
    int s = src[edge], d = dst[edge];
    float alpha = ee[edge] / denom[d];
    float4 v = ((const float4*)(xl + (size_t)s * DH))[lane];
    float* po = out + (size_t)d * DH + lane * 4;
    atomicAdd(po + 0, alpha * v.x);
    atomicAdd(po + 1, alpha * v.y);
    atomicAdd(po + 2, alpha * v.z);
    atomicAdd(po + 3, alpha * v.w);
}

// authors_to_papers segment sum: one wave per pair
__global__ void pair_scatter_kernel(const float* __restrict__ h, const int* __restrict__ paper,
                                    const int* __restrict__ author, float* __restrict__ out, int P) {
    int gid = blockIdx.x * blockDim.x + threadIdx.x;
    int p = gid >> 5, lane = gid & 31;
    if (p >= P) return;
    float4 v = ((const float4*)(h + (size_t)paper[p] * DH))[lane];
    float* po = out + (size_t)author[p] * DH + lane * 4;
    atomicAdd(po + 0, v.x);
    atomicAdd(po + 1, v.y);
    atomicAdd(po + 2, v.z);
    atomicAdd(po + 3, v.w);
}

// ---------------------------------------------------------------------------
// BatchNorm (training-mode batch stats) — one block per column
// ---------------------------------------------------------------------------
__global__ __launch_bounds__(256)
void bn_stats_kernel(const float* __restrict__ x, float* __restrict__ mean,
                     float* __restrict__ var, int n, int d) {
    __shared__ float sh[256], sh2[256];
    int col = blockIdx.x;
    float s = 0.0f, s2 = 0.0f;
    for (int i = threadIdx.x; i < n; i += 256) {
        float v = x[(size_t)i * d + col];
        s += v; s2 += v * v;
    }
    sh[threadIdx.x] = s; sh2[threadIdx.x] = s2;
    __syncthreads();
    for (int st = 128; st > 0; st >>= 1) {
        if ((int)threadIdx.x < st) { sh[threadIdx.x] += sh[threadIdx.x + st];
                                     sh2[threadIdx.x] += sh2[threadIdx.x + st]; }
        __syncthreads();
    }
    if (threadIdx.x == 0) {
        float m = sh[0] / n;
        mean[col] = m;
        var[col]  = sh2[0] / n - m * m;
    }
}

__global__ void bn_relu_kernel(float* __restrict__ x, const float* __restrict__ mean,
                               const float* __restrict__ var, const float* __restrict__ g,
                               const float* __restrict__ b, size_t n, int d) {
    size_t i = (size_t)blockIdx.x * blockDim.x + threadIdx.x;
    if (i >= n) return;
    int col = (int)(i % d);
    float v = g[col] * (x[i] - mean[col]) * rsqrtf(var[col] + 1e-5f) + b[col];
    x[i] = v > 0.0f ? v : 0.0f;
}

// ---------------------------------------------------------------------------
// LSTM pointwise (gates i,f,g,o packed as z[:, 0:128,128:256,256:384,384:512])
// ---------------------------------------------------------------------------
__global__ void lstm_point_kernel(const float* __restrict__ z, float* __restrict__ h,
                                  float* __restrict__ c, int Na) {
    size_t i = (size_t)blockIdx.x * blockDim.x + threadIdx.x;
    if (i >= (size_t)Na * DH) return;
    size_t rb = (i >> 7) * (size_t)(4 * DH);
    int col = (int)(i & (DH - 1));
    float ig = sigmoidf_(z[rb + col]);
    float fg = sigmoidf_(z[rb + DH + col]);
    float gg = tanhf(z[rb + 2 * DH + col]);
    float og = sigmoidf_(z[rb + 3 * DH + col]);
    float cc = fg * c[i] + ig * gg;
    c[i] = cc;
    h[i] = og * tanhf(cc);
}

// ---------------------------------------------------------------------------
// Final link scores: out[j*El+e] = sum_d F_j[src][d]*F_j[dst][d]*postW[j][d] + postb[j]
// F_0 = h_out, F_1..4 = heads[0..3].  One wave per edge, float4 per lane.
// ---------------------------------------------------------------------------
__global__ void link_out_kernel(const float* __restrict__ hout, const float* __restrict__ heads,
                                const int* __restrict__ src, const int* __restrict__ dst,
                                const float* __restrict__ postW, const float* __restrict__ postb,
                                float* __restrict__ out, int El, size_t headStride) {
    int gid = blockIdx.x * blockDim.x + threadIdx.x;
    int edge = gid >> 5, lane = gid & 31;
    if (edge >= El) return;
    size_t s = (size_t)src[edge] * DH, t = (size_t)dst[edge] * DH;
    for (int j = 0; j < 5; ++j) {
        const float* base = (j == 0) ? hout : heads + (size_t)(j - 1) * headStride;
        float4 a = ((const float4*)(base + s))[lane];
        float4 b = ((const float4*)(base + t))[lane];
        float4 w = ((const float4*)(postW + j * DH))[lane];
        float p = a.x * b.x * w.x + a.y * b.y * w.y + a.z * b.z * w.z + a.w * b.w * w.w;
        for (int off = 16; off > 0; off >>= 1) p += __shfl_down(p, off, 32);
        if (lane == 0) out[(size_t)j * El + edge] = p + postb[j];
    }
}

// ---------------------------------------------------------------------------
// Host-side orchestration
// ---------------------------------------------------------------------------
extern "C" void kernel_launch(void* const* d_in, const int* in_sizes, int n_in,
                              void* d_out, int out_size, void* d_ws, size_t ws_size,
                              hipStream_t stream)
{
    (void)n_in; (void)out_size; (void)ws_size;
    const float* x_c   = (const float*)d_in[0];
    const float* x_a   = (const float*)d_in[1];
    const int*   ei_c  = (const int*)d_in[2];
    const int*   ei_a  = (const int*)d_in[3];
    const int*   a2p_a = (const int*)d_in[4];
    const int*   a2p_p = (const int*)d_in[5];
    const int*   eli   = (const int*)d_in[6];
    const float* Wc_l  = (const float*)d_in[7];
    const float* Wc_r  = (const float*)d_in[8];
    const float* att_c = (const float*)d_in[9];
    const float* b_c   = (const float*)d_in[10];
    const float* Wa_l  = (const float*)d_in[11];
    const float* Wa_r  = (const float*)d_in[12];
    const float* att_a = (const float*)d_in[13];
    const float* b_a   = (const float*)d_in[14];
    const float* pre_W = (const float*)d_in[15];
    const float* pre_b = (const float*)d_in[16];
    const float* Wih   = (const float*)d_in[17];
    const float* Whh   = (const float*)d_in[18];
    const float* bih   = (const float*)d_in[19];
    const float* bhh   = (const float*)d_in[20];
    const float* rW1   = (const float*)d_in[21];
    const float* rb1   = (const float*)d_in[22];
    const float* rg1   = (const float*)d_in[23];
    const float* rbe1  = (const float*)d_in[24];
    const float* rW2   = (const float*)d_in[25];
    const float* rb2   = (const float*)d_in[26];
    const float* rg2   = (const float*)d_in[27];
    const float* rbe2  = (const float*)d_in[28];
    const float* rW3   = (const float*)d_in[29];
    const float* rb3   = (const float*)d_in[30];
    const float* hW    = (const float*)d_in[31];
    const float* hb    = (const float*)d_in[32];
    const float* postW = (const float*)d_in[33];
    const float* postb = (const float*)d_in[34];
    float* out = (float*)d_out;

    const int FA = 19, IN = DH + FA, IN_PAD = 160;   // 147 padded to multiple of 32
    const int Nc = in_sizes[0] / DH;
    const int Na = in_sizes[1] / FA;
    const int Ec = in_sizes[2] / 2;     // includes self loops
    const int Ea = in_sizes[3] / 2;
    const int P  = in_sizes[4];
    const int El = in_sizes[6] / 2;
    const int* src_c = ei_c;       const int* dst_c = ei_c + Ec;
    const int* src_a = ei_a;       const int* dst_a = ei_a + Ea;
    const int* src_l = eli;        const int* dst_l = eli + El;

    // -------- workspace arena --------
    size_t off = 0;
    auto alloc = [&](size_t bytes) -> float* {
        float* p = (float*)((char*)d_ws + off);
        off = (off + bytes + 255) & ~(size_t)255;
        return p;
    };
    const size_t ncD = (size_t)Nc * DH, naD = (size_t)Na * DH;
    float* xl     = alloc(ncD * 4);          // reused for author layers (Nc >= Na)
    float* xr     = alloc(ncD * 4);
    float* hc0    = alloc(ncD * 4);
    float* hc1    = alloc(ncD * 4);
    float* e_b    = alloc((size_t)Ec * 4);
    float* ee_b   = alloc((size_t)Ec * 4);
    float* emax   = alloc((size_t)Nc * 4);
    float* denom  = alloc((size_t)Nc * 4);
    float* collab = alloc(naD * 4);
    float* catb   = alloc((size_t)Na * IN_PAD * 4);
    float* convs  = alloc(4 * naD * 4);      // convs[t] = convs + t*naD
    float* z_b    = alloc((size_t)Na * 4 * DH * 4);
    float* hls    = alloc(naD * 4);
    float* cls    = alloc(naD * 4);
    float* heads  = alloc(4 * naD * 4);
    float* t1     = alloc(naD * 4);
    float* t2     = alloc(naD * 4);
    float* mn     = alloc(DH * 4);
    float* vr     = alloc(DH * 4);
    _Float16* pb1  = (_Float16*)alloc((size_t)IN_PAD * DH * 2);   // scratch packed weights
    _Float16* pb2  = (_Float16*)alloc((size_t)IN_PAD * DH * 2);
    _Float16* pWih = (_Float16*)alloc((size_t)DH * 4 * DH * 2);   // LSTM weights packed once
    _Float16* pWhh = (_Float16*)alloc((size_t)DH * 4 * DH * 2);

    auto cdiv = [](long long a, long long b) { return (int)((a + b - 1) / b); };

    auto packB = [&](const float* B, _Float16* dst, int Kp, int Ksrc, int N, bool transB) {
        pack_b_kernel<<<cdiv((long long)Kp * N, 256), 256, 0, stream>>>(B, dst, Kp, Ksrc, N, transB ? 1 : 0);
    };
    // only two output widths exist: N=128 and N=512 (LSTM gates)
    auto gemm = [&](const float* A, const _Float16* Bp, const float* bias, float* C,
                    int M, int N, int K, int flags) {
        dim3 g(cdiv(M, 128), N / 64);
        if (N == DH)
            wmma_gemm_kernel<DH><<<g, 256, 0, stream>>>(A, Bp, bias, C, M, K, flags);
        else
            wmma_gemm_kernel<4 * DH><<<g, 256, 0, stream>>>(A, Bp, bias, C, M, K, flags);
    };
    auto fill = [&](float* p, float v, size_t n) {
        fill_kernel<<<cdiv((long long)n, 256), 256, 0, stream>>>(p, v, n);
    };

    // one GATv2 layer + relu
    auto run_gat = [&](const float* hin, float* hout, const float* Wl, const float* Wr,
                       const float* att, const float* bias, const int* src, const int* dst,
                       int E, int Nn) {
        packB(Wl, pb1, DH, DH, DH, false);
        packB(Wr, pb2, DH, DH, DH, false);
        gemm(hin, pb1, nullptr, xl, Nn, DH, DH, 0);
        gemm(hin, pb2, nullptr, xr, Nn, DH, DH, 0);
        fill(emax, -__builtin_huge_valf(), (size_t)Nn);
        fill(denom, 0.0f, (size_t)Nn);
        edge_score_kernel<<<cdiv((long long)E * 32, 256), 256, 0, stream>>>(xl, xr, att, src, dst, e_b, E);
        edge_max_kernel<<<cdiv(E, 256), 256, 0, stream>>>(e_b, dst, emax, E);
        edge_exp_kernel<<<cdiv(E, 256), 256, 0, stream>>>(e_b, dst, emax, ee_b, denom, E);
        set_bias_rows_kernel<<<cdiv((long long)Nn * DH, 256), 256, 0, stream>>>(hout, bias, (size_t)Nn, DH);
        edge_scatter_kernel<<<cdiv((long long)E * 32, 256), 256, 0, stream>>>(xl, ee_b, denom, src, dst, hout, E);
        relu_kernel<<<cdiv((long long)Nn * DH, 256), 256, 0, stream>>>(hout, (size_t)Nn * DH);
    };

    // -------- citation GAT stack (4 layers, ping-pong) --------
    const float* hin = x_c;
    float* pong[2] = { hc0, hc1 };
    for (int i = 0; i < 4; ++i) {
        float* ho = pong[i & 1];
        run_gat(hin, ho, Wc_l + (size_t)i * DH * DH, Wc_r + (size_t)i * DH * DH,
                att_c + i * DH, b_c + i * DH, src_c, dst_c, Ec, Nc);
        hin = ho;
    }

    // -------- authors_to_papers aggregation + padded pre-linear --------
    fill(collab, 0.0f, naD);
    pair_scatter_kernel<<<cdiv((long long)P * 32, 256), 256, 0, stream>>>(hin, a2p_p, a2p_a, collab, P);
    fill(catb, 0.0f, (size_t)Na * IN_PAD);
    concat_kernel<<<cdiv((long long)Na * IN, 256), 256, 0, stream>>>(x_a, collab, catb, Na, FA, DH, IN_PAD);
    packB(pre_W, pb1, IN_PAD, IN, DH, false);     // zero-pads K 147 -> 160 in pack
    gemm(catb, pb1, pre_b, convs, Na, DH, IN_PAD, 0);   // convs[0] = x

    // -------- author GAT stack (3 layers) --------
    for (int i = 0; i < 3; ++i)
        run_gat(convs + (size_t)i * naD, convs + (size_t)(i + 1) * naD,
                Wa_l + (size_t)i * DH * DH, Wa_r + (size_t)i * DH * DH,
                att_a + i * DH, b_a + i * DH, src_a, dst_a, Ea, Na);

    // -------- LSTM over the 4 conv outputs --------
    packB(Wih, pWih, DH, DH, 4 * DH, true);       // [512,128] used as B^T, packed once
    packB(Whh, pWhh, DH, DH, 4 * DH, true);
    fill(hls, 0.0f, naD);
    fill(cls, 0.0f, naD);
    for (int t = 0; t < 4; ++t) {
        gemm(convs + (size_t)t * naD, pWih, bih, z_b, Na, 4 * DH, DH, 0);
        gemm(hls, pWhh, bhh, z_b, Na, 4 * DH, DH, GF_ACCUM);
        lstm_point_kernel<<<cdiv((long long)naD, 256), 256, 0, stream>>>(z_b, hls, cls, Na);
    }

    // -------- 4 residual heads on x = convs[3] --------
    const float* xfin = convs + 3 * naD;
    for (int i = 0; i < 4; ++i) {
        size_t wo = (size_t)i * DH * DH; int vo = i * DH;
        packB(rW1 + wo, pb1, DH, DH, DH, false);
        gemm(xfin, pb1, rb1 + vo, t1, Na, DH, DH, 0);
        bn_stats_kernel<<<DH, 256, 0, stream>>>(t1, mn, vr, Na, DH);
        bn_relu_kernel<<<cdiv((long long)naD, 256), 256, 0, stream>>>(t1, mn, vr, rg1 + vo, rbe1 + vo, naD, DH);
        packB(rW2 + wo, pb1, DH, DH, DH, false);
        gemm(t1, pb1, rb2 + vo, t2, Na, DH, DH, 0);
        bn_stats_kernel<<<DH, 256, 0, stream>>>(t2, mn, vr, Na, DH);
        bn_relu_kernel<<<cdiv((long long)naD, 256), 256, 0, stream>>>(t2, mn, vr, rg2 + vo, rbe2 + vo, naD, DH);
        add_kernel<<<cdiv((long long)naD, 256), 256, 0, stream>>>(xfin, t2, t1, naD);   // x + r
        packB(rW3 + wo, pb1, DH, DH, DH, false);
        gemm(t1, pb1, rb3 + vo, t2, Na, DH, DH, 0);
        packB(hW + wo, pb1, DH, DH, DH, false);
        gemm(t2, pb1, hb + vo, heads + (size_t)i * naD, Na, DH, DH, 0);
    }

    // -------- final 5 link scores --------
    link_out_kernel<<<cdiv((long long)El * 32, 256), 256, 0, stream>>>(
        hls, heads, src_l, dst_l, postW, postb, out, El, naD);
}
